// MaskedAttention_45449343926629
// MI455X (gfx1250) — matrix operand order repaired and checked by reading, not compiled, not run on hardware
//
#include <hip/hip_runtime.h>
#include <hip/hip_bf16.h>

typedef __bf16 bf16;
typedef __attribute__((ext_vector_type(8)))  __bf16 v8bf;
typedef __attribute__((ext_vector_type(16))) __bf16 v16bf;
typedef __attribute__((ext_vector_type(8)))  float  v8f;

#define B_   8
#define S_   1024
#define H_   1024
#define NH_  16
#define DH_  64

#define BM 128
#define BN 128
#define BK 32
#define KCHUNKS (H_ / BK)

static __device__ inline v16bf cat8(v8bf lo, v8bf hi) {
  v16bf r;
#pragma unroll
  for (int i = 0; i < 8; ++i) { r[i] = lo[i]; r[i + 8] = hi[i]; }
  return r;
}

// Async DMA of 32 bytes (2 x b128) global -> LDS; tracked by ASYNCcnt.
static __device__ inline void async_copy32(unsigned lds, const bf16* g) {
  asm volatile("global_load_async_to_lds_b128 %0, %1, off"
               :: "v"(lds), "v"(g) : "memory");
  asm volatile("global_load_async_to_lds_b128 %0, %1, off offset:16"
               :: "v"(lds + 16u), "v"(g) : "memory");
}
static __device__ inline void wait_async0() {
  asm volatile("s_wait_asynccnt 0x0" ::: "memory");
}

// ---------------------------------------------------------------------------
// One-shot fp32 -> bf16 conversion (O(N^2); keeps GEMM loops cvt-free).
// ---------------------------------------------------------------------------
__global__ void cvt_kernel(const float* __restrict__ src, bf16* __restrict__ dst,
                           int n8) {
  int i = blockIdx.x * blockDim.x + threadIdx.x;
  if (i < n8) {
    const float* s = src + (size_t)i * 8;
    v8bf o;
#pragma unroll
    for (int j = 0; j < 8; ++j) o[j] = (bf16)s[j];
    *(v8bf*)(dst + (size_t)i * 8) = o;
  }
}

// ---------------------------------------------------------------------------
// LDS-staged, async-pipelined bf16 GEMM: C = A(MxK) @ W(NxK)^T + bias.
// Block = 128x128 tile, 8 waves as 4x2 of 32x64 wave tiles (8 acc each).
// Per 32-wide k-chunk: stage A(128x32) + B(128x32) bf16 into double-buffered
// LDS with GLOBAL_LOAD_ASYNC_TO_LDS_B128 while WMMAs consume the other buffer.
// mode 0/1: bf16 out [b,h,s,d] (Q/K); mode 2: bf16 out [b,h,d,s] (V^T);
// mode 3: fp32 out row-major (final projection).
// ---------------------------------------------------------------------------
__global__ __launch_bounds__(256) void gemm_kernel(
    const bf16* __restrict__ A, const bf16* __restrict__ W,
    const float* __restrict__ bias,
    bf16* __restrict__ Ybf, float* __restrict__ Yf, int mode) {
  __shared__ bf16 lds_a[2][BM * BK];
  __shared__ bf16 lds_b[2][BN * BK];

  const int tid  = threadIdx.x;
  const int warp = tid >> 5, lane = tid & 31;
  const int m = lane & 15, g = lane >> 4;
  const int wm = warp >> 1, wn = warp & 1;          // 4 x 2 wave grid
  const int bm0 = ((int)blockIdx.x >> 3) * BM;      // 64 row blocks
  const int bn0 = ((int)blockIdx.x & 7) * BN;       // 8 col blocks

  // Cooperative copy mapping: thread -> (row = tid/2, 32-byte half = tid&1).
  const int crow = tid >> 1, chalf = tid & 1;
  const bf16* Ag = A + (size_t)(bm0 + crow) * H_ + chalf * 16;
  const bf16* Wg = W + (size_t)(bn0 + crow) * H_ + chalf * 16;
  const unsigned laBase = (unsigned)(uintptr_t)&lds_a[0][0] + crow * (BK * 2) + chalf * 32;
  const unsigned lbBase = (unsigned)(uintptr_t)&lds_b[0][0] + crow * (BK * 2) + chalf * 32;
  const unsigned BUFB = BM * BK * 2;                // 8192 bytes per buffer

  v8f acc[2][4] = {{{}, {}, {}, {}}, {{}, {}, {}, {}}};

  // Prime the pipeline: stage chunk 0 into buffer 0.
  async_copy32(laBase, Ag);
  async_copy32(lbBase, Wg);
  wait_async0();
  __syncthreads();

  for (int kc = 0; kc < KCHUNKS; ++kc) {
    const int cur = kc & 1;
    if (kc + 1 < KCHUNKS) {                         // stage next chunk (async)
      const int ko = (kc + 1) * BK;
      async_copy32(laBase + (cur ^ 1) * BUFB, Ag + ko);
      async_copy32(lbBase + (cur ^ 1) * BUFB, Wg + ko);
    }

    // Consume current chunk from LDS (layouts mirror row-major tiles).
    const bf16* la = &lds_a[cur][0];
    const bf16* lb = &lds_b[cur][0];
    v16bf a[2], b[4];
#pragma unroll
    for (int i = 0; i < 2; ++i) {
      const bf16* row = la + (wm * 32 + i * 16 + m) * BK;
      a[i] = *(const v16bf*)(row + g * 16);
    }
#pragma unroll
    for (int j = 0; j < 4; ++j)
      b[j] = *(const v16bf*)(lb + (wn * 64 + j * 16 + m) * BK + g * 16);
#pragma unroll
    for (int j = 0; j < 4; ++j) {
      acc[0][j] = __builtin_amdgcn_wmma_f32_16x16x32_bf16(false, a[0], false, b[j],
                                                          (short)0, acc[0][j], false, false);
      acc[1][j] = __builtin_amdgcn_wmma_f32_16x16x32_bf16(false, a[1], false, b[j],
                                                          (short)0, acc[1][j], false, false);
    }

    wait_async0();        // next chunk's DMA landed in LDS
    __syncthreads();      // all waves done reading cur; visibility of nxt
  }

#pragma unroll
  for (int j = 0; j < 4; ++j) {
    const int ncol = bn0 + wn * 64 + j * 16 + m;
    const int h = ncol >> 6, d = ncol & 63;
    const float bvv = bias[ncol];
#pragma unroll
    for (int i = 0; i < 2; ++i) {
#pragma unroll
      for (int r = 0; r < 8; ++r) {
        const int mg = bm0 + wm * 32 + i * 16 + r + 8 * g;   // C layout: row = r + 8g
        const float v = acc[i][j][r] + bvv;
        if (mode == 3) {
          Yf[(size_t)mg * H_ + ncol] = v;
        } else {
          const int bb = mg >> 10, s = mg & 1023;
          size_t idx;
          if (mode == 2) idx = (((size_t)(bb * NH_ + h)) * DH_ + d) * S_ + s;
          else           idx = (((size_t)(bb * NH_ + h)) * S_ + s) * DH_ + d;
          Ybf[idx] = (bf16)v;
        }
      }
    }
  }
}

// ---------------------------------------------------------------------------
// Row-wise sum of squares for the L2-attention mask term.
// ---------------------------------------------------------------------------
__global__ void sq_kernel(const bf16* __restrict__ Qb, const bf16* __restrict__ Kb,
                          float* __restrict__ q_sq, float* __restrict__ k_sq) {
  const int R = B_ * NH_ * S_;
  int i = blockIdx.x * blockDim.x + threadIdx.x;
  const bf16* src = (i < R) ? Qb : Kb;
  float* dst      = (i < R) ? q_sq : k_sq;
  int row = (i < R) ? i : i - R;
  const bf16* p = src + (size_t)row * DH_;
  float s = 0.f;
#pragma unroll
  for (int d = 0; d < DH_; ++d) { float v = (float)p[d]; s += v * v; }
  dst[row] = s;
}

// ---------------------------------------------------------------------------
// Flash-style L2 attention. One wave handles 16 queries of one (b,h):
// 4 WMMAs QK^T per 32-key block, fp32 online softmax with half-wave shuffle
// reductions, P transposed through LDS into the WMMA A layout, 4 WMMAs P@V.
// ---------------------------------------------------------------------------
__global__ void attn_kernel(const bf16* __restrict__ Qb, const bf16* __restrict__ Kb,
                            const bf16* __restrict__ Vt, const float* __restrict__ q_sq,
                            const float* __restrict__ k_sq, const float* __restrict__ mask,
                            bf16* __restrict__ Ob) {
  __shared__ bf16 ptile[16 * 32];
  const int wid = blockIdx.x;
  const int bh = wid >> 6;                 // b*NH + h
  const int qt = wid & 63;
  const int bb = bh >> 4, h = bh & 15;
  const int lane = threadIdx.x & 31;
  const int m = lane & 15, g = lane >> 4;
  const int qbase = qt * 16;
  const float scale = 0.125f, scale2 = 0.25f;   // sqrt(NH/H), 2*scale

  const bf16* Qrow = Qb + ((size_t)bh * S_ + qbase + m) * DH_;
  v16bf qa[2];
#pragma unroll
  for (int cch = 0; cch < 2; ++cch)
    qa[cch] = *(const v16bf*)(Qrow + cch * 32 + g * 16);

  float qsr[8];
#pragma unroll
  for (int r = 0; r < 8; ++r) qsr[r] = q_sq[(size_t)bh * S_ + qbase + r + 8 * g];

  float rmax[8], rsum[8];
#pragma unroll
  for (int r = 0; r < 8; ++r) { rmax[r] = -1e30f; rsum[r] = 0.f; }
  v8f acc[4] = {{}, {}, {}, {}};

  for (int kb = 0; kb < S_ / 32; ++kb) {
    const int keybase = kb * 32;

    const bf16* Kn0 = Kb + ((size_t)bh * S_ + keybase + m) * DH_;
    const bf16* Kn1 = Kn0 + (size_t)16 * DH_;
    __builtin_prefetch(Kn0 + 32 * DH_, 0, 1);
    v8f c0 = {}, c1 = {};
    c0 = __builtin_amdgcn_wmma_f32_16x16x32_bf16(false, qa[0], false,
           *(const v16bf*)(Kn0 + g * 16), (short)0, c0, false, false);
    c0 = __builtin_amdgcn_wmma_f32_16x16x32_bf16(false, qa[1], false,
           *(const v16bf*)(Kn0 + 32 + g * 16), (short)0, c0, false, false);
    c1 = __builtin_amdgcn_wmma_f32_16x16x32_bf16(false, qa[0], false,
           *(const v16bf*)(Kn1 + g * 16), (short)0, c1, false, false);
    c1 = __builtin_amdgcn_wmma_f32_16x16x32_bf16(false, qa[1], false,
           *(const v16bf*)(Kn1 + 32 + g * 16), (short)0, c1, false, false);

    float ksv0 = k_sq[(size_t)bh * S_ + keybase + m];
    float ksv1 = k_sq[(size_t)bh * S_ + keybase + 16 + m];

    float sc[2][8];
#pragma unroll
    for (int r = 0; r < 8; ++r) {
      const int row = qbase + r + 8 * g;
      sc[0][r] = scale2 * c0[r] + mask[(size_t)row * S_ + keybase + m]
               - scale * (qsr[r] + ksv0);
      sc[1][r] = scale2 * c1[r] + mask[(size_t)row * S_ + keybase + 16 + m]
               - scale * (qsr[r] + ksv1);
    }

    float p[2][8];
#pragma unroll
    for (int r = 0; r < 8; ++r) {
      float lm = fmaxf(sc[0][r], sc[1][r]);
#pragma unroll
      for (int off = 1; off < 16; off <<= 1) lm = fmaxf(lm, __shfl_xor(lm, off, 32));
      float nm = fmaxf(rmax[r], lm);
      float fr = __expf(rmax[r] - nm);
      float p0 = __expf(sc[0][r] - nm);
      float p1 = __expf(sc[1][r] - nm);
      p[0][r] = p0; p[1][r] = p1;
      float ps = p0 + p1;
#pragma unroll
      for (int off = 1; off < 16; off <<= 1) ps += __shfl_xor(ps, off, 32);
      rsum[r] = rsum[r] * fr + ps;
      rmax[r] = nm;
#pragma unroll
      for (int t = 0; t < 4; ++t) acc[t][r] *= fr;
    }

    __syncthreads();
#pragma unroll
    for (int hN = 0; hN < 2; ++hN)
#pragma unroll
      for (int r = 0; r < 8; ++r)
        ptile[(r + 8 * g) * 32 + hN * 16 + m] = (bf16)p[hN][r];
    __syncthreads();
    v16bf pa = *(const v16bf*)(ptile + m * 32 + g * 16);

    const bf16* Vbase = Vt + ((size_t)bh * DH_ + m) * S_ + keybase + g * 16;
    __builtin_prefetch(Vbase + 32, 0, 1);
#pragma unroll
    for (int t = 0; t < 4; ++t) {
      v16bf vb = *(const v16bf*)(Vbase + (size_t)t * 16 * S_);
      acc[t] = __builtin_amdgcn_wmma_f32_16x16x32_bf16(false, pa, false, vb,
                                                       (short)0, acc[t], false, false);
    }
  }

#pragma unroll
  for (int r = 0; r < 8; ++r) {
    const float inv = 1.0f / rsum[r];
    const int row = qbase + r + 8 * g;
#pragma unroll
    for (int t = 0; t < 4; ++t) {
      const int col = h * 64 + t * 16 + m;
      Ob[((size_t)bb * S_ + row) * H_ + col] = (bf16)(acc[t][r] * inv);
    }
  }
}

// ---------------------------------------------------------------------------
extern "C" void kernel_launch(void* const* d_in, const int* in_sizes, int n_in,
                              void* d_out, int out_size, void* d_ws, size_t ws_size,
                              hipStream_t stream) {
  (void)in_sizes; (void)n_in; (void)out_size; (void)ws_size;
  const float* x    = (const float*)d_in[0];
  const float* mask = (const float*)d_in[1];
  const float* Wq = (const float*)d_in[2]; const float* bq = (const float*)d_in[3];
  const float* Wk = (const float*)d_in[4]; const float* bk = (const float*)d_in[5];
  const float* Wv = (const float*)d_in[6]; const float* bv = (const float*)d_in[7];
  const float* Wo = (const float*)d_in[8]; const float* bo = (const float*)d_in[9];
  float* out = (float*)d_out;

  const size_t XE  = (size_t)B_ * S_ * H_;          // 8M elements
  const size_t WE  = (size_t)H_ * H_;               // 1M elements
  const size_t QKV = (size_t)B_ * NH_ * S_ * DH_;   // 8M elements

  bf16* Xb  = (bf16*)d_ws;
  bf16* Wqb = Xb + XE;
  bf16* Wkb = Wqb + WE;
  bf16* Wvb = Wkb + WE;
  bf16* Wob = Wvb + WE;
  bf16* Qb  = Wob + WE;
  bf16* Kb  = Qb + QKV;
  bf16* Vt  = Kb + QKV;
  bf16* Ob  = Vt + QKV;
  float* q_sq = (float*)(Ob + QKV);
  float* k_sq = q_sq + (size_t)B_ * NH_ * S_;

  // fp32 -> bf16 staging (once)
  cvt_kernel<<<4096, 256, 0, stream>>>(x, Xb, (int)(XE / 8));
  cvt_kernel<<<512, 256, 0, stream>>>(Wq, Wqb, (int)(WE / 8));
  cvt_kernel<<<512, 256, 0, stream>>>(Wk, Wkb, (int)(WE / 8));
  cvt_kernel<<<512, 256, 0, stream>>>(Wv, Wvb, (int)(WE / 8));
  cvt_kernel<<<512, 256, 0, stream>>>(Wo, Wob, (int)(WE / 8));

  // 64 x 8 blocks of 128x128, 8 waves/block
  gemm_kernel<<<512, 256, 0, stream>>>(Xb, Wqb, bq, Qb, nullptr, 0);
  gemm_kernel<<<512, 256, 0, stream>>>(Xb, Wkb, bk, Kb, nullptr, 1);
  gemm_kernel<<<512, 256, 0, stream>>>(Xb, Wvb, bv, Vt, nullptr, 2);
  sq_kernel<<<1024, 256, 0, stream>>>(Qb, Kb, q_sq, k_sq);
  attn_kernel<<<8192, 32, 0, stream>>>(Qb, Kb, Vt, q_sq, k_sq, mask, Ob);
  gemm_kernel<<<512, 256, 0, stream>>>(Ob, Wob, bo, nullptr, out, 3);
}
